// ScaleDotProductAttention_25434796327363
// MI455X (gfx1250) — matrix operand order repaired and checked by reading, not compiled, not run on hardware
//
#include <hip/hip_runtime.h>
#include <stdint.h>

// Problem constants (match reference): B=2, H=16, S=2048, D=64, causal.
#define BB 2
#define HH 16
#define SS 2048
#define DD 64

typedef __attribute__((ext_vector_type(16))) __bf16 v16bf;
typedef __attribute__((ext_vector_type(2)))  __bf16 v2bf;
typedef __attribute__((ext_vector_type(8)))  float  v8f;

union Frag { v16bf v; uint4 q[2]; unsigned u[8]; };

// fp32 pair -> packed bf16 (selects v_cvt_pk_bf16_f32)
__device__ __forceinline__ unsigned pkbf(float a, float b) {
  v2bf t;
  t.x = (__bf16)a;
  t.y = (__bf16)b;
  return __builtin_bit_cast(unsigned, t);
}

// exchange with the other 16-lane half (identity mapping), pure VALU
__device__ __forceinline__ float permx16(float v) {
  int i = __float_as_int(v);
  i = __builtin_amdgcn_permlanex16(i, i, 0x76543210, 0xfedcba98, false, false);
  return __int_as_float(i);
}

// Swizzled LDS byte offsets (XOR 16B chunks to spread banks; keeps 16B alignment).
__device__ __forceinline__ int kOff(int row, int byteoff) {  // K tile: 32 rows x 128B
  return row * 128 + (byteoff ^ ((row & 7) << 4));
}
__device__ __forceinline__ int vOff(int row, int byteoff) {  // Vt tile: 64 rows x 64B
  return row * 64 + (byteoff ^ ((row & 3) << 4));
}

// Process one 32-key tile: S^T = K.Q^T, online softmax, O^T += V^T.P^T
template <bool MASKED>
__device__ __forceinline__ void attn_tile(const unsigned char* sK,
                                          const unsigned char* sV,
                                          const Frag& q0, const Frag& q1,
                                          int kt, int wrow, int ln, int g,
                                          float& mr, float& lr, v8f o[4]) {
  // ---- scores S^T: two 16(key)x16(query) C-tiles ----
  v8f st[2];
#pragma unroll
  for (int t = 0; t < 2; ++t) {
    const int keyrow = 16 * t + ln;           // A-layout row = key
    Frag kA0, kA1;                            // d 0..31 and d 32..63
    kA0.q[0] = *(const uint4*)(sK + kOff(keyrow, 16 * g));
    kA0.q[1] = *(const uint4*)(sK + kOff(keyrow, 32 + 16 * g));
    kA1.q[0] = *(const uint4*)(sK + kOff(keyrow, 64 + 16 * g));
    kA1.q[1] = *(const uint4*)(sK + kOff(keyrow, 96 + 16 * g));
    v8f c;
#pragma unroll
    for (int j = 0; j < 8; ++j) c[j] = 0.0f;
    c = __builtin_amdgcn_wmma_f32_16x16x32_bf16(false, kA0.v, false, q0.v,
                                                (short)0, c, false, false);
    c = __builtin_amdgcn_wmma_f32_16x16x32_bf16(false, kA1.v, false, q1.v,
                                                (short)0, c, false, false);
    st[t] = c;
  }

  // ---- causal mask (boundary tiles only): key = kt+16t+8g+j, query = wrow+ln
  if (MASKED) {
#pragma unroll
    for (int t = 0; t < 2; ++t) {
      const int thr = wrow + ln - (kt + 16 * t + 8 * g);  // keep j <= thr
#pragma unroll
      for (int j = 0; j < 8; ++j)
        if (j > thr) st[t][j] = -__builtin_inff();
    }
  }

  // ---- per-query (per-lane) max: in-lane tree + one cross-half exchange ----
  float tmax = st[0][0];
#pragma unroll
  for (int j = 1; j < 8; ++j) tmax = fmaxf(tmax, st[0][j]);
#pragma unroll
  for (int j = 0; j < 8; ++j) tmax = fmaxf(tmax, st[1][j]);
  tmax = fmaxf(tmax, permx16(tmax));

  const float nm = fmaxf(mr, tmax);
  const float alpha = __builtin_amdgcn_exp2f(mr - nm);   // scores already in log2 units
  mr = nm;

  // ---- p = exp2(s - m), per-lane row sum ----
  float rs = 0.0f;
#pragma unroll
  for (int t = 0; t < 2; ++t)
#pragma unroll
    for (int j = 0; j < 8; ++j) {
      const float p = __builtin_amdgcn_exp2f(st[t][j] - nm);
      st[t][j] = p;
      rs += p;
    }
  rs += permx16(rs);
  lr = lr * alpha + rs;

  // ---- rescale O^T accumulator (alpha is a per-lane scalar) ----
#pragma unroll
  for (int dt = 0; dt < 4; ++dt)
#pragma unroll
    for (int j = 0; j < 8; ++j) o[dt][j] *= alpha;

  // ---- build P^T B-fragment in registers (no LDS round trip) ----
  // lane (g) needs keys [16g, 16g+16) of query column ln:
  //   g=0: lo = own st0 (keys 0-7),  hi = sibling st0 (keys 8-15)
  //   g=1: lo = sibling st1 (16-23), hi = own st1 (keys 24-31)
  float lo[8], hi[8];
#pragma unroll
  for (int j = 0; j < 8; ++j) {
    const float x0 = permx16(st[0][j]);
    const float x1 = permx16(st[1][j]);
    lo[j] = g ? x1 : st[0][j];
    hi[j] = g ? st[1][j] : x0;
  }
  Frag pB;
#pragma unroll
  for (int j = 0; j < 4; ++j) {
    pB.u[j]     = pkbf(lo[2 * j], lo[2 * j + 1]);
    pB.u[4 + j] = pkbf(hi[2 * j], hi[2 * j + 1]);
  }

  // ---- O^T += V^T . P^T  (4 d-tiles of 16) ----
#pragma unroll
  for (int dt = 0; dt < 4; ++dt) {
    const int drow = 16 * dt + ln;            // A-layout row = d
    Frag vA;
    vA.q[0] = *(const uint4*)(sV + vOff(drow, 32 * g));
    vA.q[1] = *(const uint4*)(sV + vOff(drow, 32 * g + 16));
    o[dt] = __builtin_amdgcn_wmma_f32_16x16x32_bf16(false, vA.v, false, pB.v,
                                                    (short)0, o[dt], false, false);
  }
}

__global__ __launch_bounds__(256)
void fattn_causal_bf16_wmma(const float* __restrict__ Q,
                            const float* __restrict__ K,
                            const float* __restrict__ V,
                            float* __restrict__ O) {
  __shared__ __align__(16) unsigned char sK[32 * 128];   // K tile: 32 keys x 64 bf16
  __shared__ __align__(16) unsigned char sV[64 * 64];    // V^T tile: 64 d x 32 bf16

  const int tid  = threadIdx.x;
  const int wave = tid >> 5;                  // 0..7
  const int lane = tid & 31;
  const int g    = lane >> 4;
  const int ln   = lane & 15;

  const int qb = blockIdx.x * 128;            // 128 query rows per block
  const int bh = blockIdx.y;
  const size_t base = (size_t)bh * SS * DD;
  const int wrow = qb + wave * 16;

  // ---- persistent Q^T B-fragments, pre-scaled by (1/sqrt(D))*log2(e) ----
  const float qscale = 0.125f * 1.44269504088896340736f;
  Frag qB[2];
  {
    const float* qp = Q + base + (size_t)(wrow + ln) * DD;
#pragma unroll
    for (int f = 0; f < 2; ++f) {
      const float4* p4 = (const float4*)(qp + 32 * f + 8 * g);
      const float4 a = p4[0], b = p4[1], c = p4[4], d = p4[5];
      qB[f].u[0] = pkbf(a.x * qscale, a.y * qscale);
      qB[f].u[1] = pkbf(a.z * qscale, a.w * qscale);
      qB[f].u[2] = pkbf(b.x * qscale, b.y * qscale);
      qB[f].u[3] = pkbf(b.z * qscale, b.w * qscale);
      qB[f].u[4] = pkbf(c.x * qscale, c.y * qscale);
      qB[f].u[5] = pkbf(c.z * qscale, c.w * qscale);
      qB[f].u[6] = pkbf(d.x * qscale, d.y * qscale);
      qB[f].u[7] = pkbf(d.z * qscale, d.w * qscale);
    }
  }

  v8f o[4];
#pragma unroll
  for (int dt = 0; dt < 4; ++dt)
#pragma unroll
    for (int j = 0; j < 8; ++j) o[dt][j] = 0.0f;
  float mr = -__builtin_inff();
  float lr = 0.0f;

  const int kend = qb + 128;                  // causal bound for this q block
  for (int kt = 0; kt < kend; kt += 32) {
    __syncthreads();

    // ---- stage K tile: row-major bf16 (256 threads -> 8 floats each) ----
    {
      const int krow = tid >> 3;              // 0..31
      const int dseg = (tid & 7) * 8;         // 0,8,...,56
      const float4* kp = (const float4*)(K + base + (size_t)(kt + krow) * DD + dseg);
      const float4 a = kp[0], b = kp[1];
      uint4 u0;
      u0.x = pkbf(a.x, a.y); u0.y = pkbf(a.z, a.w);
      u0.z = pkbf(b.x, b.y); u0.w = pkbf(b.z, b.w);
      *(uint4*)(sK + kOff(krow, dseg * 2)) = u0;
    }
    // ---- stage V^T tile: per-thread d-column, 8 coalesced dword loads,
    //      one b128 LDS store ----
    {
      const int d  = tid & 63;
      const int kq = tid >> 6;                // key quarter: 8kq .. 8kq+7
      const float* vp = V + base + (size_t)(kt + 8 * kq) * DD + d;
      float vv[8];
#pragma unroll
      for (int i = 0; i < 8; ++i) vv[i] = vp[(size_t)i * DD];
      uint4 u0;
      u0.x = pkbf(vv[0], vv[1]); u0.y = pkbf(vv[2], vv[3]);
      u0.z = pkbf(vv[4], vv[5]); u0.w = pkbf(vv[6], vv[7]);
      *(uint4*)(sV + vOff(d, 16 * kq)) = u0;
    }
    __syncthreads();

    // ---- wave-uniform tile classification: skip / unmasked / masked ----
    if (kt <= wrow + 15) {
      if (kt + 31 <= wrow)
        attn_tile<false>(sK, sV, qB[0], qB[1], kt, wrow, ln, g, mr, lr, o);
      else
        attn_tile<true>(sK, sV, qB[0], qB[1], kt, wrow, ln, g, mr, lr, o);
    }
  }

  // ---- epilogue: normalize, store fp32 (lane owns query row wrow+ln) ----
  const float rl = 1.0f / lr;
  float* Op = O + base + (size_t)(wrow + ln) * DD;
#pragma unroll
  for (int dt = 0; dt < 4; ++dt) {
    float4 r0, r1;
    r0.x = o[dt][0] * rl; r0.y = o[dt][1] * rl;
    r0.z = o[dt][2] * rl; r0.w = o[dt][3] * rl;
    r1.x = o[dt][4] * rl; r1.y = o[dt][5] * rl;
    r1.z = o[dt][6] * rl; r1.w = o[dt][7] * rl;
    *(float4*)(Op + 16 * dt + 8 * g)     = r0;
    *(float4*)(Op + 16 * dt + 8 * g + 4) = r1;
  }
}

extern "C" void kernel_launch(void* const* d_in, const int* in_sizes, int n_in,
                              void* d_out, int out_size, void* d_ws, size_t ws_size,
                              hipStream_t stream) {
  (void)in_sizes; (void)n_in; (void)out_size; (void)d_ws; (void)ws_size;
  const float* q = (const float*)d_in[0];
  const float* k = (const float*)d_in[1];
  const float* v = (const float*)d_in[2];
  // d_in[3] is the causal mask; causality is applied analytically in-kernel.
  float* out = (float*)d_out;
  dim3 grid(SS / 128, BB * HH);
  fattn_causal_bf16_wmma<<<grid, 256, 0, stream>>>(q, k, v, out);
}
